// Multi_Relation_30580167147773
// MI455X (gfx1250) — matrix operand order repaired and checked by reading, not compiled, not run on hardware
//
#include <hip/hip_runtime.h>
#include <hip/hip_bf16.h>

// ---------------------------------------------------------------------------
// MI455X / gfx1250: wave32, WMMA bf16 16x16x32 (f32 accumulate).
// Problem: N=1024 proposals, D=128, H=16 heads, DK=64, DG=64, DV=8.
// Memory floor = 256MB geo stream / 23.3TB/s ~= 11us; GEMMs run in bf16 WMMA
// to stay far under that floor. All heavy data paths use contiguous 16B/lane
// fragment loads matching the ISA's WMMA VGPR layouts.
// ---------------------------------------------------------------------------

typedef __attribute__((ext_vector_type(16))) __bf16 v16bf;
typedef __attribute__((ext_vector_type(8)))  float  v8f;

union Frag16 {
    v16bf v;
    uint4 q[2];
};

#define WMMA_BF16(A, B, C) \
    __builtin_amdgcn_wmma_f32_16x16x32_bf16(false, (A), false, (B), (short)0, (C), false, false)

// ---------------------------------------------------------------------------
// Kernel 1: K,Q projections -> bf16 rows [h][n][64].  A row staged in LDS,
// weight reads coalesced over k.  (Small GEMM, L2-resident; VALU is fine.)
// ---------------------------------------------------------------------------
__global__ void proj_kq_kernel(const float* __restrict__ App,
                               const float* __restrict__ W_K, const float* __restrict__ b_K,
                               const float* __restrict__ W_Q, const float* __restrict__ b_Q,
                               __bf16* __restrict__ Kb, __bf16* __restrict__ Qb) {
    const int n = blockIdx.x, h = blockIdx.y, t = threadIdx.x;
    __shared__ float sA[128];
    sA[t] = App[n * 128 + t];
    __syncthreads();
    const bool isK = (t < 64);
    const int  k   = isK ? t : (t - 64);
    const float* W = isK ? W_K : W_Q;
    float acc = isK ? b_K[h * 64 + k] : b_Q[h * 64 + k];
#pragma unroll 8
    for (int d = 0; d < 128; ++d)
        acc += sA[d] * W[(h * 128 + d) * 64 + k];
    __bf16* dst = isK ? Kb : Qb;
    dst[((size_t)h * 1024 + n) * 64 + k] = (__bf16)acc;
}

// ---------------------------------------------------------------------------
// Kernel 2: V projection, stored v-major f32: Vt[h][v][m].  16M MACs, tiny;
// appearance (512KB) is L2-resident so access order is uncritical.
// ---------------------------------------------------------------------------
__global__ void proj_v_kernel(const float* __restrict__ App,
                              const float* __restrict__ W_V, const float* __restrict__ b_V,
                              float* __restrict__ Vt) {
    const int idx = blockIdx.x * 256 + threadIdx.x;          // 131072 = 16*8*1024
    const int m = idx & 1023, v = (idx >> 10) & 7, h = idx >> 13;
    float acc = b_V[h * 8 + v];
    for (int d = 0; d < 128; ++d)
        acc += App[m * 128 + d] * W_V[(h * 128 + d) * 8 + v];
    Vt[(h * 8 + v) * 1024 + m] = acc;
}

// ---------------------------------------------------------------------------
// Kernel 3 (heavy): one wave per 16x16 (n,m) tile.
//  pass A: geo gate via pair-GEMM  [16 pairs x 64g] x [64g x 16 heads]  (WMMA)
//          -> relu -> LDS sWG[n][m][h] (padded 17 -> bank-conflict-free)
//  pass B: per head: w_A = K Qt (2x WMMA, K=64), scale 1/8, combine
//          w_nom = gate * exp(w_A)  -> bf16, tiled layout [h][nt][mt][n*16+m].
// geo fp32 is converted to bf16 in-register while streaming (bandwidth-bound).
// ---------------------------------------------------------------------------
__global__ void wnom_kernel(const __bf16* __restrict__ Kb, const __bf16* __restrict__ Qb,
                            const float* __restrict__ Geo,
                            const float* __restrict__ W_G, const float* __restrict__ b_G,
                            __bf16* __restrict__ WN) {
    const int mt = blockIdx.x, nt = blockIdx.y;
    const int lane = threadIdx.x;
    const int lo = lane & 15, hi = lane >> 4;

    __shared__ float sWG[16][16][17];   // [n_local][m_local][head], padded

    // B operand for the gate GEMM: W_G^T (64x16), bf16.  lane=head col,
// elems 0..7 = K hi*8.., 8..15 = K 16+hi*8.. within each 32-chunk.
    Frag16 wg0, wg1;
#pragma unroll
    for (int j = 0; j < 16; ++j) {
        wg0.v[j] = (__bf16)W_G[lo * 64 +      hi * 8 + j + (j & 8)];
        wg1.v[j] = (__bf16)W_G[lo * 64 + 32 + hi * 8 + j + (j & 8)];
    }

    // ---- pass A: geometric gate, one subtile per n_local t ----
    for (int t = 0; t < 16; ++t) {
        // A row = pair (n = nt*16+t, m = mt*16+lo); elems 0..7 = g base, 8..15 = g base+16
        const float* gp = Geo + ((size_t)(nt * 16 + t) * 1024 + mt * 16 + lo) * 64;
        Frag16 ga0, ga1;
        const int ab = hi * 8;
#pragma unroll
        for (int j = 0; j < 8; ++j) {
            ga0.v[j]     = (__bf16)gp[ab + j];
            ga0.v[8 + j] = (__bf16)gp[ab + 16 + j];
            ga1.v[j]     = (__bf16)gp[ab + 32 + j];
            ga1.v[8 + j] = (__bf16)gp[ab + 48 + j];
        }
        v8f acc;
        const float bg = b_G[lo];               // N (col) = head = lo
#pragma unroll
        for (int r = 0; r < 8; ++r) acc[r] = bg;
        acc = WMMA_BF16(ga0.v, wg0.v, acc);
        acc = WMMA_BF16(ga1.v, wg1.v, acc);
#pragma unroll
        for (int r = 0; r < 8; ++r) {           // D: M = r + hi*8 = m_local, N = lo = head
            const float g = acc[r] > 0.f ? acc[r] : 0.f;
            sWG[t][r + hi * 8][lo] = g;
        }
    }
    __syncthreads();

    // ---- pass B: w_A per head (K=64 -> 2 WMMA), fuse exp + write w_nom ----
    for (int h = 0; h < 16; ++h) {
        const __bf16* kr = Kb + (((size_t)h * 1024) + nt * 16 + lo) * 64;  // A row n = lo
        const __bf16* qr = Qb + (((size_t)h * 1024) + mt * 16 + lo) * 64;  // B col m = lo
        Frag16 ka0, ka1, qb0, qb1;
        const int ab = hi * 8;    // A: elems 0..7 = K ab.., 8..15 = K ab+16..
        ka0.q[0] = *(const uint4*)(kr + ab);       ka0.q[1] = *(const uint4*)(kr + ab + 16);
        ka1.q[0] = *(const uint4*)(kr + ab + 32);  ka1.q[1] = *(const uint4*)(kr + ab + 48);
        const int bb = hi * 8;    // B: elems 0..7 = K bb.., 8..15 = K bb+16..
        qb0.q[0] = *(const uint4*)(qr + bb);       qb0.q[1] = *(const uint4*)(qr + bb + 16);
        qb1.q[0] = *(const uint4*)(qr + bb + 32);  qb1.q[1] = *(const uint4*)(qr + bb + 48);
        v8f acc = {};
        acc = WMMA_BF16(ka0.v, qb0.v, acc);
        acc = WMMA_BF16(ka1.v, qb1.v, acc);
        // D: (n = r + hi*8, m = lo).  w_nom = gate * exp(w_A / 8)
        __bf16* dst = WN + ((((size_t)h * 64 + nt) * 64 + mt) * 256);
#pragma unroll
        for (int r = 0; r < 8; ++r) {
            const int n = r + hi * 8;
            const float wn = sWG[n][lo][h] * __expf(acc[r] * 0.125f);
            dst[n * 16 + lo] = (__bf16)wn;       // lanes -> contiguous m: coalesced
        }
    }
}

// ---------------------------------------------------------------------------
// Kernel 4: deterministic column sums  colsum[h][m] = sum_n w_nom[h][n][m].
// One block per (h, mtile); fully coalesced reads of the tiled layout.
// ---------------------------------------------------------------------------
__global__ void colsum_kernel(const __bf16* __restrict__ WN, float* __restrict__ CS) {
    const int h = blockIdx.x >> 6, mt = blockIdx.x & 63;
    const int ml = threadIdx.x & 15, np = threadIdx.x >> 4;
    float acc = 0.f;
    for (int nt = 0; nt < 64; ++nt)
        acc += (float)WN[(((size_t)h * 64 + nt) * 64 + mt) * 256 + np * 16 + ml];
    __shared__ float red[16][17];
    red[np][ml] = acc;
    __syncthreads();
    if (np == 0) {
        float s = 0.f;
#pragma unroll
        for (int j = 0; j < 16; ++j) s += red[j][ml];
        CS[h * 1024 + mt * 16 + ml] = s;
    }
}

// ---------------------------------------------------------------------------
// Kernel 5: fold 1/colsum into V, emit bf16 WMMA B operand VSB[h][16][1024]
// (rows 8..15 zero so the 16-wide B has 8 valid value columns).
// ---------------------------------------------------------------------------
__global__ void vsb_kernel(const float* __restrict__ Vt, const float* __restrict__ CS,
                           __bf16* __restrict__ VSB) {
    const int idx = blockIdx.x * 256 + threadIdx.x;          // 262144 = 16*16*1024
    const int m = idx & 1023, r = (idx >> 10) & 15, h = idx >> 14;
    float val = 0.f;
    if (r < 8) val = Vt[(h * 8 + r) * 1024 + m] / CS[h * 1024 + m];
    VSB[((size_t)(h * 16 + r)) * 1024 + m] = (__bf16)val;
}

// ---------------------------------------------------------------------------
// Kernel 6: f_R = w_nom x Vs per head.  One wave per (head, ntile); 32-deep
// bf16 WMMA chain over K=1024 (A tiles come straight from the tiled w_nom
// layout as 16B/lane loads).  Epilogue: second = f_R + appearance.
// ---------------------------------------------------------------------------
__global__ void fr_kernel(const __bf16* __restrict__ WN, const __bf16* __restrict__ VSB,
                          const float* __restrict__ App, float* __restrict__ Sec) {
    const int nt = blockIdx.x, h = blockIdx.y;
    const int lane = threadIdx.x, lo = lane & 15, hi = lane >> 4;
    const __bf16* vb    = VSB + ((size_t)(h * 16 + lo)) * 1024;   // B col = v
    const __bf16* wtile = WN + (((size_t)h * 64 + nt) * 64) * 256;
    v8f acc = {};
    for (int c = 0; c < 32; ++c) {                // K chunk = m = c*32 .. c*32+31
        Frag16 af, bf;
        const __bf16* a0 = wtile + (size_t)(2 * c    ) * 256 + lo * 16 + hi * 8;
        const __bf16* a1 = wtile + (size_t)(2 * c + 1) * 256 + lo * 16 + hi * 8;
        af.q[0] = *(const uint4*)a0;              // K elems 0..7
        af.q[1] = *(const uint4*)a1;              // K elems 8..15 (= +16 in chunk)
        const __bf16* bp = vb + c * 32 + hi * 8;
        bf.q[0] = *(const uint4*)bp;
        bf.q[1] = *(const uint4*)(bp + 16);
        acc = WMMA_BF16(af.v, bf.v, acc);
    }
    if (lo < 8) {                                 // D: n = r + hi*8, v = lo
#pragma unroll
        for (int r = 0; r < 8; ++r) {
            const int n = nt * 16 + r + hi * 8;
            const int o = n * 128 + h * 8 + lo;
            Sec[o] = acc[r] + App[o];
        }
    }
}

// ---------------------------------------------------------------------------
// Kernel 7: out = s0 * sigmoid(second @ W_S + b_S).  One wave per row.
// ---------------------------------------------------------------------------
__global__ void final_kernel(const float* __restrict__ Sec, const float* __restrict__ W_S,
                             const float* __restrict__ b_S, const float* __restrict__ S0,
                             float* __restrict__ Out) {
    const int lane = threadIdx.x & 31;
    const int row  = blockIdx.x * 8 + (threadIdx.x >> 5);
    float s = 0.f;
#pragma unroll
    for (int j = 0; j < 4; ++j)
        s += Sec[row * 128 + lane * 4 + j] * W_S[lane * 4 + j];
#pragma unroll
    for (int off = 16; off > 0; off >>= 1)
        s += __shfl_xor(s, off, 32);
    if (lane == 0) {
        const float s1 = 1.f / (1.f + __expf(-(s + b_S[0])));
        Out[row] = S0[row] * s1;
    }
}

// ---------------------------------------------------------------------------
extern "C" void kernel_launch(void* const* d_in, const int* in_sizes, int n_in,
                              void* d_out, int out_size, void* d_ws, size_t ws_size,
                              hipStream_t stream) {
    (void)in_sizes; (void)n_in; (void)out_size; (void)ws_size;
    const float* App = (const float*)d_in[0];
    const float* Geo = (const float*)d_in[1];
    const float* S0  = (const float*)d_in[2];
    const float* W_V = (const float*)d_in[3];
    const float* b_V = (const float*)d_in[4];
    const float* W_K = (const float*)d_in[5];
    const float* b_K = (const float*)d_in[6];
    const float* W_Q = (const float*)d_in[7];
    const float* b_Q = (const float*)d_in[8];
    const float* W_G = (const float*)d_in[9];
    const float* b_G = (const float*)d_in[10];
    const float* W_S = (const float*)d_in[11];
    const float* b_S = (const float*)d_in[12];
    float* Out = (float*)d_out;
    char*  ws  = (char*)d_ws;

    // workspace layout (~41 MB total)
    __bf16* Kb  = (__bf16*)(ws + (0ull  << 20));  // 2 MB  [16][1024][64] bf16
    __bf16* Qb  = (__bf16*)(ws + (2ull  << 20));  // 2 MB
    float*  Vt  = (float*) (ws + (4ull  << 20));  // 512 KB [16][8][1024] f32
    __bf16* VSB = (__bf16*)(ws + (5ull  << 20));  // 512 KB [16][16][1024] bf16
    float*  CS  = (float*) (ws + (6ull  << 20));  // 64 KB  [16][1024] f32
    __bf16* WN  = (__bf16*)(ws + (7ull  << 20));  // 32 MB  tiled [16][64][64][256] bf16
    float*  Sec = (float*) (ws + (40ull << 20));  // 512 KB [1024][128] f32

    proj_kq_kernel<<<dim3(1024, 16), 128, 0, stream>>>(App, W_K, b_K, W_Q, b_Q, Kb, Qb);
    proj_v_kernel <<<512, 256, 0, stream>>>(App, W_V, b_V, Vt);
    wnom_kernel   <<<dim3(64, 64), 32, 0, stream>>>(Kb, Qb, Geo, W_G, b_G, WN);
    colsum_kernel <<<1024, 256, 0, stream>>>(WN, CS);
    vsb_kernel    <<<1024, 256, 0, stream>>>(Vt, CS, VSB);
    fr_kernel     <<<dim3(64, 16), 32, 0, stream>>>(WN, VSB, App, Sec);
    final_kernel  <<<128, 256, 0, stream>>>(Sec, W_S, b_S, S0, Out);
}
